// SCNet_16922171146708
// MI455X (gfx1250) — matrix operand order repaired and checked by reading, not compiled, not run on hardware
//
#include <hip/hip_runtime.h>
#include <hip/hip_bf16.h>

typedef _Float16 half_t;
typedef _Float16 v16h __attribute__((ext_vector_type(16)));
typedef _Float16 v8h  __attribute__((ext_vector_type(8)));
typedef float    v8f  __attribute__((ext_vector_type(8)));

#define S0 3600
#define NPTS 200000

#if defined(__HIP_DEVICE_COMPILE__) && \
    __has_builtin(__builtin_amdgcn_global_load_async_to_lds_b32) && \
    __has_builtin(__builtin_amdgcn_global_load_async_to_lds_b8) && \
    __has_builtin(__builtin_amdgcn_s_wait_asynccnt)
#define HAVE_ASYNC_LDS 1
typedef __attribute__((address_space(1))) int  as1_int;
typedef __attribute__((address_space(3))) int  as3_int;
typedef __attribute__((address_space(1))) char as1_char;
typedef __attribute__((address_space(3))) char as3_char;
#else
#define HAVE_ASYNC_LDS 0
#endif

__device__ __forceinline__ float eluf(float x) {
    return x > 0.f ? x : (__expf(x) - 1.f);
}

// ---------------------------------------------------------------------------
// Scatter sparse (coord, feature) pairs onto the dense grid + mask.
// ---------------------------------------------------------------------------
__global__ void k_scatter(const int* __restrict__ coords,
                          const float* __restrict__ feats,
                          float* __restrict__ grid,
                          unsigned char* __restrict__ mask, int P) {
    int i = blockIdx.x * blockDim.x + threadIdx.x;
    if (i >= P) return;
    int r = coords[2 * i], c = coords[2 * i + 1];
    atomicAdd(&grid[(size_t)r * S0 + c], feats[i]);
    mask[(size_t)r * S0 + c] = 1;
}

// ---------------------------------------------------------------------------
// Fused: conv1 (5x5, 1->10) + ELU + sparse 5x5 maxpool (3600 -> 720).
// One block = 16x16 pooled cells; input region 84^2 staged in LDS via the
// gfx1250 async global->LDS path (fallback: synchronous loads).
// Output act1 is channel-PADDED to 16 (zeros in ch 10..15) for WMMA layer 2.
// ---------------------------------------------------------------------------
__global__ void k_conv1_pool(const float* __restrict__ grid,
                             const unsigned char* __restrict__ mask,
                             const float* __restrict__ w1,
                             half_t* __restrict__ act1,
                             unsigned char* __restrict__ mask1) {
    __shared__ float         s_tile[84 * 84];
    __shared__ unsigned char s_mask[84 * 84];
    __shared__ float         s_w[256];

    const int t  = threadIdx.x;
    const int bx = blockIdx.x, by = blockIdx.y;
    const int oy = by * 80 - 2, ox = bx * 80 - 2;

#if HAVE_ASYNC_LDS
    for (int idx = t; idx < 84 * 84; idx += 256) {
        int ly = idx / 84, lx = idx % 84;
        int gy = oy + ly, gx = ox + lx;
        if (gy >= 0 && gy < S0 && gx >= 0 && gx < S0) {
            size_t g = (size_t)gy * S0 + gx;
            __builtin_amdgcn_global_load_async_to_lds_b32(
                (as1_int*)(grid + g), (as3_int*)&s_tile[idx], 0, 0);
            __builtin_amdgcn_global_load_async_to_lds_b8(
                (as1_char*)(mask + g), (as3_char*)&s_mask[idx], 0, 0);
        } else {
            s_tile[idx] = 0.f; s_mask[idx] = 0;
        }
    }
    __builtin_amdgcn_s_wait_asynccnt(0);   // own-wave asyncs done before barrier
#else
    for (int idx = t; idx < 84 * 84; idx += 256) {
        int ly = idx / 84, lx = idx % 84;
        int gy = oy + ly, gx = ox + lx;
        float v = 0.f; unsigned char m = 0;
        if (gy >= 0 && gy < S0 && gx >= 0 && gx < S0) {
            size_t g = (size_t)gy * S0 + gx;
            v = grid[g]; m = mask[g];
        }
        s_tile[idx] = v; s_mask[idx] = m;
    }
#endif
    if (t < 250) s_w[t] = w1[t];   // w1[(ky*5+kx)*10 + c]
    __syncthreads();

    const int ty = t >> 4, tx = t & 15;        // pooled cell in tile
    float best[10];
#pragma unroll
    for (int c = 0; c < 10; ++c) best[c] = -1e30f;
    bool any = false;

    for (int wy = 0; wy < 5; ++wy) {
        for (int wx = 0; wx < 5; ++wx) {
            int ly = ty * 5 + wy + 2, lx = tx * 5 + wx + 2;
            if (!s_mask[ly * 84 + lx]) continue;
            any = true;
            float acc[10];
#pragma unroll
            for (int c = 0; c < 10; ++c) acc[c] = 0.f;
            for (int dy = 0; dy < 5; ++dy)
                for (int dx = 0; dx < 5; ++dx) {
                    float v = s_tile[(ly + dy - 2) * 84 + (lx + dx - 2)];
                    const float* wr = &s_w[(dy * 5 + dx) * 10];
#pragma unroll
                    for (int c = 0; c < 10; ++c) acc[c] += v * wr[c];
                }
#pragma unroll
            for (int c = 0; c < 10; ++c) best[c] = fmaxf(best[c], eluf(acc[c]));
        }
    }

    int pgy = by * 16 + ty, pgx = bx * 16 + tx;
    size_t pp = (size_t)pgy * 720 + pgx;
#pragma unroll
    for (int c = 0; c < 16; ++c)
        act1[pp * 16 + c] = (half_t)((c < 10 && any) ? best[c] : 0.f);
    mask1[pp] = any ? 1 : 0;
}

// ---------------------------------------------------------------------------
// Pre-pack conv weights (HWIO f32) into f16 B-fragment layout.
// wpk[(((nt*CHUNKS + chunk)*32) + lane)*16 + i] with the ISA 32x16 B layout:
// lane L -> col N = L%16; half i -> K = chunk*32 + (i&7) + (i>=8 ? 16 : 0) + (L>=16 ? 8 : 0).
// K runs over the CINP-padded im2col axis: tap = K/CINP, cin = K%CINP.
// ---------------------------------------------------------------------------
__global__ void k_pack_w(const float* __restrict__ w, half_t* __restrict__ wpk,
                         int CIN, int CINP, int COUT, int CHUNKS) {
    int idx = blockIdx.x * blockDim.x + threadIdx.x;
    int total = (COUT / 16) * CHUNKS * 512;
    if (idx >= total) return;
    int i     = idx & 15;
    int lane  = (idx >> 4) & 31;
    int chunk = (idx >> 9) % CHUNKS;
    int nt    = idx / (CHUNKS << 9);
    int kk  = chunk * 32 + (i & 7) + (i >= 8 ? 16 : 0) + (lane >= 16 ? 8 : 0);
    int tap = kk / CINP, cin = kk % CINP;
    half_t v = (half_t)0;
    if (tap < 25 && cin < CIN)
        v = (half_t)w[(size_t)(tap * CIN + cin) * COUT + nt * 16 + (lane & 15)];
    wpk[idx] = v;
}

// ---------------------------------------------------------------------------
// Implicit-GEMM 5x5 SAME conv via v_wmma_f32_16x16x32_f16.
// One wave = 16 pixels x 16 out-channels; K = 25*CINP (CINP power of two).
// A: each lane's two 8-half K-runs lie inside one tap -> one b128 load each.
// B: one 32-byte load from the pre-packed fragment buffer.
// R compile-time -> all /R, %R become magic-multiply constants.
// Output = mask ? elu(acc) : 0, f16 [pix][COUT].
// ---------------------------------------------------------------------------
template <int R, int CINP, int COUT, int CHUNKS>
__global__ void k_conv_wmma(const half_t* __restrict__ actin,
                            const unsigned char* __restrict__ maskin,
                            const half_t* __restrict__ wpk,
                            half_t* __restrict__ convout) {
    constexpr int npix = R * R;
    constexpr int MT   = (npix + 15) / 16;
    constexpr int NT   = COUT / 16;

    const int lane = threadIdx.x & 31;
    const int wid  = blockIdx.x * (blockDim.x >> 5) + (threadIdx.x >> 5);
    if (wid >= MT * NT) return;

    const int mt = wid / NT, nt = wid % NT;
    const int pbase = mt * 16, nbase = nt * 16;
    const bool hi = lane >= 16;
    const int  lm = lane & 15;

    const int  pix    = pbase + lm;
    const bool pvalid = pix < npix;
    const int  py = pix / R;           // compile-time R -> no hw division
    const int  px = pix % R;

    v8f acc = {};

    for (int chunk = 0; chunk < CHUNKS; ++chunk) {
        const int k0 = chunk * 32;
        __builtin_prefetch(&wpk[(((size_t)nt * CHUNKS + chunk + 1) * 32) * 16], 0, 1);

        // ---- A fragment: two contiguous 8-channel runs (b128 each)
        union { v16h v; v8h h[2]; } au;
#pragma unroll
        for (int rsel = 0; rsel < 2; ++rsel) {
            int kk  = k0 + (hi ? 8 : 0) + rsel * 16;
            int tap = kk / CINP;           // CINP is a power of two -> shift
            int cin = kk & (CINP - 1);
            v8h run = {};
            if (pvalid && tap < 25) {
                int iy = py + tap / 5 - 2, ix = px + tap % 5 - 2;
                if (iy >= 0 && iy < R && ix >= 0 && ix < R)
                    run = *(const v8h*)(actin + ((size_t)iy * R + ix) * CINP + cin);
            }
            au.h[rsel] = run;
        }

        // ---- B fragment: pre-packed, one 32-byte contiguous load
        const v16h b = *(const v16h*)(wpk + (((size_t)nt * CHUNKS + chunk) * 32 + lane) * 16);

        acc = __builtin_amdgcn_wmma_f32_16x16x32_f16(
            /*neg_a=*/false, au.v, /*neg_b=*/false, b,
            /*c_mod=*/(short)0, acc, /*reuse_a=*/false, /*reuse_b=*/false);
    }

    // C/D layout: VGPR r, lane L -> M = r + (L>=16 ? 8 : 0), N = L%16
#pragma unroll
    for (int r = 0; r < 8; ++r) {
        int m = r + (hi ? 8 : 0);
        int p = pbase + m;
        if (p < npix) {
            float o = maskin[p] ? eluf(acc[r]) : 0.f;
            convout[(size_t)p * COUT + nbase + lm] = (half_t)o;
        }
    }
}

// ---------------------------------------------------------------------------
// Sparse maxpool K x K (stride K): max over ACTIVE input cells, else 0.
// One thread per (pooled pixel, 8-channel group); 16-byte vector loads.
// ---------------------------------------------------------------------------
template <int RIN, int KP, int C>
__global__ void k_pool8(const half_t* __restrict__ conv,
                        const unsigned char* __restrict__ maskin,
                        half_t* __restrict__ actout,
                        unsigned char* __restrict__ maskout) {
    constexpr int Rout   = RIN / KP;
    constexpr int groups = C / 8;
    constexpr int total  = Rout * Rout * groups;

    int idx = blockIdx.x * blockDim.x + threadIdx.x;
    if (idx >= total) return;
    int g = idx % groups;
    int p = idx / groups;
    int pyo = p / Rout, pxo = p % Rout;
    float best[8];
#pragma unroll
    for (int j = 0; j < 8; ++j) best[j] = -1e30f;
    bool any = false;
#pragma unroll
    for (int wy = 0; wy < KP; ++wy)
#pragma unroll
        for (int wx = 0; wx < KP; ++wx) {
            int ip = (pyo * KP + wy) * RIN + (pxo * KP + wx);
            if (maskin[ip]) {
                any = true;
                v8h v = *(const v8h*)(conv + (size_t)ip * C + g * 8);
#pragma unroll
                for (int j = 0; j < 8; ++j) best[j] = fmaxf(best[j], (float)v[j]);
            }
        }
    v8h o;
#pragma unroll
    for (int j = 0; j < 8; ++j) o[j] = (half_t)(any ? best[j] : 0.f);
    *(v8h*)(actout + (size_t)p * C + g * 8) = o;
    if (g == 0) maskout[p] = any ? 1 : 0;
}

// ---------------------------------------------------------------------------
// FC head: NCHW flatten (f = c*81 + p) -> 20736x32 FC + ELU -> 32x5 FC
// -> softmax. One workgroup.
// ---------------------------------------------------------------------------
__global__ void k_head(const half_t* __restrict__ act4,
                       const float* __restrict__ fc1w,
                       const float* __restrict__ fc1b,
                       const float* __restrict__ fc2w,
                       const float* __restrict__ fc2b,
                       float* __restrict__ out) {
    __shared__ float part[256];
    __shared__ float h[32];
    __shared__ float lg[5];
    const int t = threadIdx.x;
    const int j = t & 31, s = t >> 5;     // 8 K-slices of 2592 each

    float sum = 0.f;
    for (int f = s * 2592; f < (s + 1) * 2592; ++f) {
        int c = f / 81, p = f % 81;       // NCHW flatten
        sum += (float)act4[(size_t)p * 256 + c] * fc1w[(size_t)f * 32 + j];
    }
    part[t] = sum;
    __syncthreads();

    if (t < 32) {
        float v = fc1b[t];
        for (int ss = 0; ss < 8; ++ss) v += part[ss * 32 + t];
        h[t] = eluf(v);
    }
    __syncthreads();

    if (t < 5) {
        float v = fc2b[t];
        for (int jj = 0; jj < 32; ++jj) v += h[jj] * fc2w[jj * 5 + t];
        lg[t] = v;
    }
    __syncthreads();

    if (t == 0) {
        float mx = lg[0];
        for (int i = 1; i < 5; ++i) mx = fmaxf(mx, lg[i]);
        float e[5], den = 0.f;
        for (int i = 0; i < 5; ++i) { e[i] = __expf(lg[i] - mx); den += e[i]; }
        for (int i = 0; i < 5; ++i) out[i] = e[i] / den;
    }
}

// ---------------------------------------------------------------------------
extern "C" void kernel_launch(void* const* d_in, const int* in_sizes, int n_in,
                              void* d_out, int out_size, void* d_ws, size_t ws_size,
                              hipStream_t stream) {
    const int*   coords = (const int*)  d_in[0];
    const float* feats  = (const float*)d_in[1];
    const float* w1     = (const float*)d_in[2];
    const float* w2     = (const float*)d_in[3];
    const float* w3     = (const float*)d_in[4];
    const float* w4     = (const float*)d_in[5];
    const float* fc1w   = (const float*)d_in[6];
    const float* fc1b   = (const float*)d_in[7];
    const float* fc2w   = (const float*)d_in[8];
    const float* fc2b   = (const float*)d_in[9];
    float* out = (float*)d_out;
    (void)in_sizes; (void)n_in; (void)out_size; (void)ws_size;

    char* ws = (char*)d_ws;
    size_t off = 0;
    auto suballoc = [&](size_t bytes) -> char* {
        char* p = ws + off;
        off = (off + bytes + 255) & ~(size_t)255;
        return p;
    };

    float*         grid0  = (float*)        suballoc((size_t)S0 * S0 * 4);
    unsigned char* mask0  = (unsigned char*)suballoc((size_t)S0 * S0);
    half_t*        act1   = (half_t*)       suballoc((size_t)720 * 720 * 16 * 2); // padded to 16ch
    unsigned char* mask1  = (unsigned char*)suballoc((size_t)720 * 720);
    half_t*        conv2o = (half_t*)       suballoc((size_t)720 * 720 * 64 * 2);
    half_t*        act2   = (half_t*)       suballoc((size_t)144 * 144 * 64 * 2);
    unsigned char* mask2  = (unsigned char*)suballoc((size_t)144 * 144);
    half_t*        conv3o = (half_t*)       suballoc((size_t)144 * 144 * 128 * 2);
    half_t*        act3   = (half_t*)       suballoc((size_t)36 * 36 * 128 * 2);
    unsigned char* mask3  = (unsigned char*)suballoc((size_t)36 * 36);
    half_t*        conv4o = (half_t*)       suballoc((size_t)36 * 36 * 256 * 2);
    half_t*        act4   = (half_t*)       suballoc((size_t)81 * 256 * 2);
    unsigned char* mask4  = (unsigned char*)suballoc((size_t)81);
    half_t*        wpk2   = (half_t*)       suballoc((size_t)4  * 13  * 512 * 2);
    half_t*        wpk3   = (half_t*)       suballoc((size_t)8  * 50  * 512 * 2);
    half_t*        wpk4   = (half_t*)       suballoc((size_t)16 * 100 * 512 * 2);

    // grid/mask are accumulated into by scatter -> must be re-zeroed per call
    (void)hipMemsetAsync(grid0, 0, (size_t)S0 * S0 * 4, stream);
    (void)hipMemsetAsync(mask0, 0, (size_t)S0 * S0, stream);

    // weight fragment packing (tiny)
    k_pack_w<<<(4  * 13  * 512 + 255) / 256, 256, 0, stream>>>(w2, wpk2, 10,  16,  64,  13);
    k_pack_w<<<(8  * 50  * 512 + 255) / 256, 256, 0, stream>>>(w3, wpk3, 64,  64,  128, 50);
    k_pack_w<<<(16 * 100 * 512 + 255) / 256, 256, 0, stream>>>(w4, wpk4, 128, 128, 256, 100);

    k_scatter<<<(NPTS + 255) / 256, 256, 0, stream>>>(coords, feats, grid0, mask0, NPTS);

    // L1: conv 1->10 + ELU + pool5  (3600 -> 720), writes 16ch-padded act1
    k_conv1_pool<<<dim3(45, 45), 256, 0, stream>>>(grid0, mask0, w1, act1, mask1);

    // L2: conv 10->64 @720 (WMMA, CINP=16), then pool5 -> 144
    {
        constexpr int waves = ((720 * 720 + 15) / 16) * (64 / 16);
        k_conv_wmma<720, 16, 64, 13><<<(waves + 7) / 8, 256, 0, stream>>>(act1, mask1, wpk2, conv2o);
        constexpr int total = 144 * 144 * (64 / 8);
        k_pool8<720, 5, 64><<<(total + 255) / 256, 256, 0, stream>>>(conv2o, mask1, act2, mask2);
    }
    // L3: conv 64->128 @144 (WMMA), then pool4 -> 36
    {
        constexpr int waves = ((144 * 144 + 15) / 16) * (128 / 16);
        k_conv_wmma<144, 64, 128, 50><<<(waves + 7) / 8, 256, 0, stream>>>(act2, mask2, wpk3, conv3o);
        constexpr int total = 36 * 36 * (128 / 8);
        k_pool8<144, 4, 128><<<(total + 255) / 256, 256, 0, stream>>>(conv3o, mask2, act3, mask3);
    }
    // L4: conv 128->256 @36 (WMMA), then pool4 -> 9
    {
        constexpr int waves = ((36 * 36 + 15) / 16) * (256 / 16);
        k_conv_wmma<36, 128, 256, 100><<<(waves + 7) / 8, 256, 0, stream>>>(act3, mask3, wpk4, conv4o);
        constexpr int total = 9 * 9 * (256 / 8);
        k_pool8<36, 4, 256><<<(total + 255) / 256, 256, 0, stream>>>(conv4o, mask3, act4, mask4);
    }
    // FC head + softmax
    k_head<<<1, 256, 0, stream>>>(act4, fc1w, fc1b, fc2w, fc2b, out);
}
